// MultiHeadAttention_81020263072017
// MI455X (gfx1250) — matrix operand order repaired and checked by reading, not compiled, not run on hardware
//
#include <hip/hip_runtime.h>
#include <hip/hip_bf16.h>

// ---------------------------------------------------------------------------
// MultiHeadAttention for MI455X (gfx1250): bf16 WMMA everywhere.
// cvt(f32->bf16) -> GEMM(Q,K,V) [double-buffered, async-to-LDS] ->
// flash-attention (transposed S/O: per-lane softmax state, 1 shuffle per
// reduction) -> GEMM(out, f32).
// ---------------------------------------------------------------------------

#define B_   4
#define L_   2048
#define D_   1024
#define H_   16
#define HD_  64
#define ML_  (B_ * L_)          // 8192 rows
#define NEGV (-10000000.0f)
#define SCALE 0.125f            // 1/sqrt(64)

typedef __attribute__((ext_vector_type(16))) __bf16 v16bf;
typedef __attribute__((ext_vector_type(8)))  float  v8f;

__device__ __forceinline__ v8f wmma_bf16(v16bf a, v16bf b, v8f c) {
  return __builtin_amdgcn_wmma_f32_16x16x32_bf16(false, a, false, b,
                                                 (short)0, c, false, false);
}

__device__ __forceinline__ __bf16 f2bf(float f) {
  unsigned u = __float_as_uint(f);
  unsigned r = (u + 0x7FFFu + ((u >> 16) & 1u)) >> 16;
  unsigned short s = (unsigned short)r;
  return __builtin_bit_cast(__bf16, s);
}

struct alignas(32) frag_bits { uint4 lo, hi; };

__device__ __forceinline__ v16bf frag_from(uint4 lo, uint4 hi) {
  frag_bits b{lo, hi};
  return __builtin_bit_cast(v16bf, b);
}

// 16 contiguous bf16 (32B, 16B-aligned) -> B-side fragment
// (elements ordered k = 16*half + 0..15).
__device__ __forceinline__ v16bf ld_frag_contig(const __bf16* p) {
  return frag_from(*reinterpret_cast<const uint4*>(p),
                   *reinterpret_cast<const uint4*>(p + 8));
}

// A-layout fragment: lane m = lane&15 owns row m; elements ordered
// k = 16*half + 0..15 -> chunks at +16*half and +16*half+8.
__device__ __forceinline__ v16bf ld_frag_a(const __bf16* row, int half) {
  return frag_from(*reinterpret_cast<const uint4*>(row + 16 * half),
                   *reinterpret_cast<const uint4*>(row + 16 * half + 8));
}

// ---- gfx1250 async global->LDS copy (ASYNCcnt path), guarded --------------
#if defined(__AMDGCN__) &&                                                     \
    __has_builtin(__builtin_amdgcn_global_load_async_to_lds_b128) &&           \
    __has_builtin(__builtin_amdgcn_s_wait_asynccnt)
#define HAVE_ASYNC_LDS 1
#else
#define HAVE_ASYNC_LDS 0
#endif

typedef int v4i_ __attribute__((vector_size(16)));

__device__ __forceinline__ void copy_b128_g2l(const __bf16* g, __bf16* l) {
#if HAVE_ASYNC_LDS
  __builtin_amdgcn_global_load_async_to_lds_b128(
      (__attribute__((address_space(1))) v4i_*)(uintptr_t)g,
      (__attribute__((address_space(3))) v4i_*)(unsigned)(uintptr_t)l, 0, 0);
#else
  *reinterpret_cast<uint4*>(l) = *reinterpret_cast<const uint4*>(g);
#endif
}

__device__ __forceinline__ void async_wait_all() {
#if HAVE_ASYNC_LDS
  __builtin_amdgcn_s_wait_asynccnt(0);
#endif
}

// ---------------------------------------------------------------------------
// Elementwise f32 -> bf16 convert (4-wide)
// ---------------------------------------------------------------------------
__global__ __launch_bounds__(256) void cvt_f32_bf16(const float* __restrict__ in,
                                                    __bf16* __restrict__ out,
                                                    int n4) {
  int i = blockIdx.x * 256 + threadIdx.x;
  if (i < n4) {
    float4 v = reinterpret_cast<const float4*>(in)[i];
    __bf16* o = out + i * 4;
    o[0] = f2bf(v.x); o[1] = f2bf(v.y); o[2] = f2bf(v.z); o[3] = f2bf(v.w);
  }
}

// ---------------------------------------------------------------------------
// bf16 GEMM: C[M,N] = A[M,K] * B[K,N]. Block tile 128x128, 8 waves of 32x64.
// Double-buffered LDS; A tile staged via async-to-LDS; B staged transposed.
// Row strides padded to 40 bf16 (80B) for bank spread, 16B-aligned.
// ---------------------------------------------------------------------------
#define GLD 40
template <bool OUT_BF16>
__global__ __launch_bounds__(256) void gemm_bf16_kernel(
    const __bf16* __restrict__ A, const __bf16* __restrict__ Bm,
    void* __restrict__ C, int M, int N, int K) {
  __shared__ __align__(16) __bf16 sA[2][128 * GLD];    // [m][k]
  __shared__ __align__(16) __bf16 sBt[2][128 * GLD];   // [n][k] (transposed)

  const int tid  = threadIdx.x;
  const int lane = tid & 31;
  const int wave = tid >> 5;
  const int n0 = blockIdx.x * 128;
  const int m0 = blockIdx.y * 128;
  const int wm = (wave >> 1) * 32;   // 4 waves along M
  const int wn = (wave & 1)  * 64;   // 2 waves along N
  const int half = lane >> 4, nn = lane & 15;

  v8f acc[2][4] = {};

  auto stage = [&](int buf, int kb) {
    // A tile 128x32 row-major: pure copy -> async DMA to LDS
    for (int c = tid; c < 512; c += 256) {
      int r = c >> 2, co = (c & 3) * 8;
      copy_b128_g2l(&A[(size_t)(m0 + r) * K + kb + co],
                    &sA[buf][r * GLD + co]);
    }
    // B tile 32x128 transposed -> sBt[n][k]
    for (int c = tid; c < 512; c += 256) {
      int k = c >> 4, nc = (c & 15) * 8;
      uint4 gv =
          *reinterpret_cast<const uint4*>(&Bm[(size_t)(kb + k) * N + n0 + nc]);
      __bf16 t[8];
      *reinterpret_cast<uint4*>(t) = gv;
#pragma unroll
      for (int j = 0; j < 8; ++j) sBt[buf][(nc + j) * GLD + k] = t[j];
    }
  };

  stage(0, 0);
  async_wait_all();
  __syncthreads();

  int buf = 0;
  for (int kb = 0; kb < K; kb += 32) {
    const bool has_next = (kb + 32 < K);
    if (has_next) stage(buf ^ 1, kb + 32);  // overlaps with this step's math

    v16bf a[2], b[4];
#pragma unroll
    for (int mi = 0; mi < 2; ++mi)
      a[mi] = ld_frag_a(&sA[buf][(wm + mi * 16 + nn) * GLD], half);
#pragma unroll
    for (int ni = 0; ni < 4; ++ni)
      b[ni] = ld_frag_contig(&sBt[buf][(wn + ni * 16 + nn) * GLD + 16 * half]);

#pragma unroll
    for (int mi = 0; mi < 2; ++mi)
#pragma unroll
      for (int ni = 0; ni < 4; ++ni)
        acc[mi][ni] = wmma_bf16(a[mi], b[ni], acc[mi][ni]);

    if (has_next) async_wait_all();
    __syncthreads();
    buf ^= 1;
  }

#pragma unroll
  for (int mi = 0; mi < 2; ++mi)
#pragma unroll
    for (int ni = 0; ni < 4; ++ni)
#pragma unroll
      for (int r = 0; r < 8; ++r) {
        int row = m0 + wm + mi * 16 + r + 8 * half;
        int col = n0 + wn + ni * 16 + nn;
        float v = acc[mi][ni][r];
        if (OUT_BF16)
          reinterpret_cast<__bf16*>(C)[(size_t)row * N + col] = f2bf(v);
        else
          reinterpret_cast<float*>(C)[(size_t)row * N + col] = v;
      }
}

// ---------------------------------------------------------------------------
// Flash attention, TRANSPOSED form. Block = 128 queries of one (batch,head);
// 8 waves each own 16 queries; 32-key iterations.
//   S^T = K * Q^T   (A = K tile rows, B = Q^T; C cols = queries)
//   => each lane (q = lane&15) holds 8 of its 16 scores per fragment;
//      softmax state (m, l) is a per-lane scalar; halves merged with ONE
//      __shfl_xor(..., 16, 32) instead of a 4-step butterfly per row.
//   O^T = V^T * P^T (A = V^T from transposed LDS, B = P^T scratch)
//   => alpha rescale of O is a per-lane scalar multiply.
// Strict causal mask (diagonal masked), fill = -1e7.
// ---------------------------------------------------------------------------
#define SKLD 72   // sK row stride (bf16): 144B, 16B-aligned
#define SVLD 40   // sVt row stride
#define SPLD 40   // sPt row stride
__global__ __launch_bounds__(256) void attn_kernel(
    const __bf16* __restrict__ Q, const __bf16* __restrict__ Kg,
    const __bf16* __restrict__ V, __bf16* __restrict__ O) {
  __shared__ __align__(16) __bf16 sK[32 * SKLD];      // [key][d]
  __shared__ __align__(16) __bf16 sVt[64 * SVLD];     // [d][key] (transposed)
  __shared__ __align__(16) __bf16 sPt[8][16 * SPLD];  // [q][key] per wave

  const int tid  = threadIdx.x;
  const int lane = tid & 31;
  const int wave = tid >> 5;
  const int q0 = blockIdx.x * 128;
  const int bh = blockIdx.y;
  const int bb = bh >> 4, h = bh & 15;
  const int rowbase = bb * L_;
  const int colbase = h * HD_;
  const int qw = q0 + wave * 16;
  const int m = lane & 15, half = lane >> 4;
  const int qn = qw + m;              // the query this lane tracks

  // Q^T B-fragments: lane = query column, contiguous d run of 16 from global
  v16bf qb[2];
#pragma unroll
  for (int kc = 0; kc < 2; ++kc)
    qb[kc] = ld_frag_contig(
        &Q[(size_t)(rowbase + qn) * D_ + colbase + kc * 32 + 16 * half]);

  float mrun = NEGV, lrun = 0.0f;
  v8f oT[4] = {};  // O^T frags: rows = d (per dc chunk), cols = queries

  const int nkt = q0 / 32 + 4;  // keys [0, q0+128)
  for (int kt = 0; kt < nkt; ++kt) {
    const int kbase = kt * 32;
    {
      int r = tid >> 3, co = (tid & 7) * 8;
      // K tile row-major: pure copy -> async DMA to LDS
      copy_b128_g2l(&Kg[(size_t)(rowbase + kbase + r) * D_ + colbase + co],
                    &sK[r * SKLD + co]);
      // V tile transposed -> sVt[d][key]
      uint4 gv = *reinterpret_cast<const uint4*>(
          &V[(size_t)(rowbase + kbase + r) * D_ + colbase + co]);
      __bf16 t[8];
      *reinterpret_cast<uint4*>(t) = gv;
#pragma unroll
      for (int j = 0; j < 8; ++j) sVt[(co + j) * SVLD + r] = t[j];
      if (kt + 1 < nkt)
        __builtin_prefetch(
            &V[(size_t)(rowbase + kbase + 32 + r) * D_ + colbase], 0, 1);
    }
    async_wait_all();
    __syncthreads();

    // S^T = K * Q^T: A frags = K rows (keys), contiguous d; 2 key halves x
    // 2 d-chunks -> 4 WMMAs into two C frags (keys 0-15 / 16-31 as rows).
    v16bf ka[2][2];
#pragma unroll
    for (int t = 0; t < 2; ++t)
#pragma unroll
      for (int kc = 0; kc < 2; ++kc)
        ka[t][kc] = ld_frag_a(&sK[(t * 16 + m) * SKLD + kc * 32], half);
    v8f s0 = {}, s1 = {};
    s0 = wmma_bf16(ka[0][0], qb[0], s0);
    s0 = wmma_bf16(ka[0][1], qb[1], s0);
    s1 = wmma_bf16(ka[1][0], qb[0], s1);
    s1 = wmma_bf16(ka[1][1], qb[1], s1);

    // mask + per-lane online softmax. C layout: reg r = key row r + 8*half,
    // col = lane&15 = query.
    float vals[16];
#pragma unroll
    for (int r = 0; r < 8; ++r) {
      int key0 = kbase + r + 8 * half;
      int key1 = key0 + 16;
      vals[r]     = (key0 < qn) ? s0[r] * SCALE : NEGV;  // strict: diag masked
      vals[8 + r] = (key1 < qn) ? s1[r] * SCALE : NEGV;
    }
    float tmax = vals[0];
#pragma unroll
    for (int j = 1; j < 16; ++j) tmax = fmaxf(tmax, vals[j]);
    tmax = fmaxf(tmax, __shfl_xor(tmax, 16, 32));  // merge key halves
    float nm = fmaxf(mrun, tmax);
    float al = __expf(mrun - nm);
    float rs = 0.0f;
    __bf16* prow = &sPt[wave][m * SPLD];
#pragma unroll
    for (int r = 0; r < 8; ++r) {
      float p0 = __expf(vals[r] - nm);
      float p1 = __expf(vals[8 + r] - nm);
      rs += p0 + p1;
      prow[r + 8 * half]      = f2bf(p0);   // key index r + 8*half
      prow[16 + r + 8 * half] = f2bf(p1);   // key index 16 + r + 8*half
    }
    rs += __shfl_xor(rs, 16, 32);
    lrun = lrun * al + rs;
    mrun = nm;

    // rescale running O^T by per-lane alpha
#pragma unroll
    for (int f = 0; f < 4; ++f)
#pragma unroll
      for (int r = 0; r < 8; ++r) oT[f][r] *= al;

    // O^T += V^T * P^T : A = V^T (lane = d row, contiguous keys),
    //                    B = P^T (lane = query col, contiguous keys).
    v16bf pb = ld_frag_contig(&sPt[wave][m * SPLD + 16 * half]);
#pragma unroll
    for (int dc = 0; dc < 4; ++dc) {
      v16bf va = ld_frag_a(&sVt[(dc * 16 + m) * SVLD], half);
      oT[dc] = wmma_bf16(va, pb, oT[dc]);
    }
    __syncthreads();
  }

  // normalize (per-lane 1/l) and write bf16 attention output [B*L, 1024].
  // O^T layout: row = d = dc*16 + r + 8*half, col = query = lane&15.
  const float inv = 1.0f / lrun;
#pragma unroll
  for (int dc = 0; dc < 4; ++dc)
#pragma unroll
    for (int r = 0; r < 8; ++r) {
      float val = oT[dc][r] * inv;
      int col = colbase + dc * 16 + r + 8 * half;
      O[(size_t)(rowbase + qn) * D_ + col] = f2bf(val);
    }
}

// ---------------------------------------------------------------------------
extern "C" void kernel_launch(void* const* d_in, const int* in_sizes, int n_in,
                              void* d_out, int out_size, void* d_ws,
                              size_t ws_size, hipStream_t stream) {
  (void)in_sizes; (void)n_in; (void)out_size; (void)ws_size;

  const float* x  = (const float*)d_in[0];
  const float* Wq = (const float*)d_in[1];
  const float* Wk = (const float*)d_in[2];
  const float* Wv = (const float*)d_in[3];
  const float* Wo = (const float*)d_in[4];

  char* ws = (char*)d_ws;
  const size_t XB = (size_t)ML_ * D_ * 2;   // 16 MB
  const size_t WB = (size_t)D_ * D_ * 2;    // 2 MB
  __bf16* xb  = (__bf16*)(ws);
  __bf16* wqb = (__bf16*)(ws + XB);
  __bf16* wkb = (__bf16*)(ws + XB + WB);
  __bf16* wvb = (__bf16*)(ws + XB + 2 * WB);
  __bf16* wob = (__bf16*)(ws + XB + 3 * WB);
  __bf16* Qb  = (__bf16*)(ws + XB + 4 * WB);
  __bf16* Kb  = (__bf16*)(ws + 2 * XB + 4 * WB);
  __bf16* Vb  = (__bf16*)(ws + 3 * XB + 4 * WB);
  __bf16* Ab  = (__bf16*)(ws + 4 * XB + 4 * WB);

  const int nx4 = ML_ * D_ / 4;
  const int nw4 = D_ * D_ / 4;
  cvt_f32_bf16<<<(nx4 + 255) / 256, 256, 0, stream>>>(x,  xb,  nx4);
  cvt_f32_bf16<<<(nw4 + 255) / 256, 256, 0, stream>>>(Wq, wqb, nw4);
  cvt_f32_bf16<<<(nw4 + 255) / 256, 256, 0, stream>>>(Wk, wkb, nw4);
  cvt_f32_bf16<<<(nw4 + 255) / 256, 256, 0, stream>>>(Wv, wvb, nw4);
  cvt_f32_bf16<<<(nw4 + 255) / 256, 256, 0, stream>>>(Wo, wob, nw4);

  dim3 ggrid(D_ / 128, ML_ / 128);  // (8, 64)
  gemm_bf16_kernel<true><<<ggrid, 256, 0, stream>>>(xb, wqb, Qb, ML_, D_, D_);
  gemm_bf16_kernel<true><<<ggrid, 256, 0, stream>>>(xb, wkb, Kb, ML_, D_, D_);
  gemm_bf16_kernel<true><<<ggrid, 256, 0, stream>>>(xb, wvb, Vb, ML_, D_, D_);

  dim3 agrid(L_ / 128, B_ * H_);    // (16, 64)
  attn_kernel<<<agrid, 256, 0, stream>>>(Qb, Kb, Vb, Ab);

  gemm_bf16_kernel<false><<<ggrid, 256, 0, stream>>>(Ab, wob, d_out, ML_, D_, D_);
}
